// AttentionProlongationGNN_64295660421656
// MI455X (gfx1250) — matrix operand order repaired and checked by reading, not compiled, not run on hardware
//
#include <hip/hip_runtime.h>

#define N_NODES 50000
#define N_EDGES 800000
#define HID 256
#define NHEAD 8
#define NLAYER 3

typedef __attribute__((ext_vector_type(16))) _Float16 v16h;
typedef __attribute__((ext_vector_type(8)))  float    v8f;
typedef __attribute__((ext_vector_type(4)))  int      v4i;

#ifndef __has_builtin
#define __has_builtin(x) 0
#endif

#if __has_builtin(__builtin_amdgcn_global_load_async_to_lds_b128)
#define HAVE_ASYNC_LDS 1
#else
#define HAVE_ASYNC_LDS 0
#endif

struct Pack32 { uint4 lo, hi; };
__device__ __forceinline__ v16h frag16(uint4 lo, uint4 hi) {
  Pack32 p{lo, hi};
  return __builtin_bit_cast(v16h, p);
}

#if HAVE_ASYNC_LDS
typedef __attribute__((address_space(1))) v4i GAv4i;   // global
typedef __attribute__((address_space(3))) v4i LAv4i;   // LDS
__device__ __forceinline__ void async_copy16(const _Float16* g, _Float16* l) {
  __builtin_amdgcn_global_load_async_to_lds_b128((GAv4i*)g, (LAv4i*)l, 0, 0);
}
__device__ __forceinline__ void wait_async() {
#if __has_builtin(__builtin_amdgcn_s_wait_asynccnt)
  __builtin_amdgcn_s_wait_asynccnt(0);
#else
  asm volatile("s_wait_asynccnt 0x0" ::: "memory");
#endif
}
#endif

// monotonic float<->uint encoding so atomicMax(uint) == max(float)
__device__ __forceinline__ unsigned encF(float f) {
  unsigned u = __float_as_uint(f);
  return (u >> 31) ? ~u : (u | 0x80000000u);
}
__device__ __forceinline__ float decF(unsigned e) {
  return __uint_as_float((e & 0x80000000u) ? (e & 0x7fffffffu) : ~e);
}

__device__ __forceinline__ void atomAddF(float* a, float v) {
  __hip_atomic_fetch_add(a, v, __ATOMIC_RELAXED, __HIP_MEMORY_SCOPE_AGENT);
}

// ---------------------------------------------------------------------------
// Tiled fp16 WMMA GEMM: C(MxNc,f32) = A(MxKd,f16) @ B(KdxNc,f16) + bias
// block = 256 threads (8 wave32), tile 128x64, K-step 32, wave patch 32x32.
// Double-buffered LDS; A tile staged with GLOBAL_LOAD_ASYNC_TO_LDS_B128 when
// available; LDS rows padded to 40 halfs (80B) for conflict-free frag reads.
// ---------------------------------------------------------------------------
template<int ACT, bool DUAL>
__global__ __launch_bounds__(256) void k_gemm(
    const _Float16* __restrict__ A, const _Float16* __restrict__ B,
    const float* __restrict__ bias, float* __restrict__ C,
    _Float16* __restrict__ C2, int M, int Kd, int Nc, int ldc2)
{
  constexpr int LDA = 40, LDB = 40;
  __shared__ __align__(16) _Float16 As[2][128 * LDA];
  __shared__ __align__(16) _Float16 Bs[2][64 * LDB];   // transposed: [n][k]
  const int tid = threadIdx.x;
  const int w   = tid >> 5, ln = tid & 31;
  const int wr  = w >> 1,  wc = w & 1;
  const int bm0 = blockIdx.x * 128;
  const int bn0 = blockIdx.y * 64;

  v8f acc[2][2] = {};

  const int arow = tid >> 1;
  const int acol = (tid & 1) * 16;
  int garow = bm0 + arow; if (garow >= M) garow = M - 1;   // clamp, keep EXEC full
  const _Float16* aptr = A + (size_t)garow * Kd + acol;
  const int aoff = arow * LDA + acol;

  const int bk  = tid >> 3;
  const int bn8 = (tid & 7) * 8;
  const _Float16* bptr = B + (size_t)bk * Nc + bn0 + bn8;

  const int khalf = (ln >> 4) * 8;    // A frag K-half select
  const int kb    = (ln >> 4) * 8;    // B frag K-half select
  const int l15   = ln & 15;
  const int nt    = Kd >> 5;

  union BReg { uint4 v; _Float16 h[8]; };

  // ---- prologue: stage tile 0 into buffer 0
  {
#if HAVE_ASYNC_LDS
    async_copy16(aptr,     &As[0][aoff]);
    async_copy16(aptr + 8, &As[0][aoff + 8]);
#else
    uint4 a0 = ((const uint4*)aptr)[0];
    uint4 a1 = ((const uint4*)aptr)[1];
    *(uint4*)&As[0][aoff]     = a0;
    *(uint4*)&As[0][aoff + 8] = a1;
#endif
    BReg bu; bu.v = *(const uint4*)bptr;
    #pragma unroll
    for (int j = 0; j < 8; ++j) Bs[0][(bn8 + j) * LDB + bk] = bu.h[j];
#if HAVE_ASYNC_LDS
    wait_async();
#endif
  }
  __syncthreads();

  for (int t = 0; t < nt; ++t) {
    const int cur = t & 1, nxt = cur ^ 1;
    const bool more = (t + 1 < nt);
    BReg bu;
#if !HAVE_ASYNC_LDS
    uint4 a0, a1;
#endif
    if (more) {
      const int k1 = (t + 1) << 5;
      bu.v = *(const uint4*)(bptr + (size_t)k1 * Nc);
#if HAVE_ASYNC_LDS
      async_copy16(aptr + k1,     &As[nxt][aoff]);       // ASYNCcnt path
      async_copy16(aptr + k1 + 8, &As[nxt][aoff + 8]);
#else
      a0 = ((const uint4*)(aptr + k1))[0];
      a1 = ((const uint4*)(aptr + k1))[1];
#endif
    }

    // ---- compute tile t from buffer `cur`
    const _Float16* as_ = As[cur];
    const _Float16* bs_ = Bs[cur];
    const int mA0 = (wr * 32      + l15) * LDA;
    const int mA1 = (wr * 32 + 16 + l15) * LDA;
    const int nB0 = (wc * 32      + l15) * LDB;
    const int nB1 = (wc * 32 + 16 + l15) * LDB;
    v16h af0 = frag16(*(const uint4*)&as_[mA0 + khalf], *(const uint4*)&as_[mA0 + 16 + khalf]);
    v16h af1 = frag16(*(const uint4*)&as_[mA1 + khalf], *(const uint4*)&as_[mA1 + 16 + khalf]);
    v16h bf0 = frag16(*(const uint4*)&bs_[nB0 + kb],    *(const uint4*)&bs_[nB0 + kb + 16]);
    v16h bf1 = frag16(*(const uint4*)&bs_[nB1 + kb],    *(const uint4*)&bs_[nB1 + kb + 16]);
    acc[0][0] = __builtin_amdgcn_wmma_f32_16x16x32_f16(false, af0, false, bf0, (short)0, acc[0][0], false, false);
    acc[0][1] = __builtin_amdgcn_wmma_f32_16x16x32_f16(false, af0, false, bf1, (short)0, acc[0][1], false, false);
    acc[1][0] = __builtin_amdgcn_wmma_f32_16x16x32_f16(false, af1, false, bf0, (short)0, acc[1][0], false, false);
    acc[1][1] = __builtin_amdgcn_wmma_f32_16x16x32_f16(false, af1, false, bf1, (short)0, acc[1][1], false, false);

    // ---- stage tile t+1 into buffer `nxt`
    if (more) {
      #pragma unroll
      for (int j = 0; j < 8; ++j) Bs[nxt][(bn8 + j) * LDB + bk] = bu.h[j];
#if HAVE_ASYNC_LDS
      wait_async();
#else
      *(uint4*)&As[nxt][aoff]     = a0;
      *(uint4*)&As[nxt][aoff + 8] = a1;
#endif
    }
    __syncthreads();
  }

  #pragma unroll
  for (int mi = 0; mi < 2; ++mi) {
    #pragma unroll
    for (int ni = 0; ni < 2; ++ni) {
      const int colg = bn0 + wc * 32 + ni * 16 + l15;
      const float bb = bias ? bias[colg] : 0.f;
      const int rbase = bm0 + wr * 32 + mi * 16 + ((ln >> 4) * 8);
      #pragma unroll
      for (int i = 0; i < 8; ++i) {
        const int rg = rbase + i;
        if (rg < M) {
          float v = acc[mi][ni][i] + bb;
          if (ACT == 1) v = fmaxf(v, 0.f);
          C[(size_t)rg * Nc + colg] = v;
          if (DUAL) C2[(size_t)rg * ldc2 + colg] = (_Float16)v;
        }
      }
    }
  }
}

// ---------------------------------------------------------------------------
// small helper kernels
// ---------------------------------------------------------------------------
__global__ void k_cvt(const float* __restrict__ in, _Float16* __restrict__ out, int n) {
  int i = blockIdx.x * 256 + threadIdx.x;
  if (i < n) out[i] = (_Float16)in[i];
}

__global__ void k_zero(float* __restrict__ p, int n) {
  int i = blockIdx.x * 256 + threadIdx.x;
  if (i < n) p[i] = 0.f;
}

__global__ void k_init_small(float* __restrict__ asum, unsigned* __restrict__ gmax, int n) {
  int i = blockIdx.x * 256 + threadIdx.x;
  if (i < n) asum[i] = 0.f;
  if (i == 0) *gmax = 0u;   // encoded -inf
}

// h = relu(x @ W_in + b_in); write f32 + f16 mirror + concat first half
__global__ void k_inproj(const float* __restrict__ x, const float* __restrict__ W,
                         const float* __restrict__ b, float* __restrict__ h,
                         _Float16* __restrict__ h16, _Float16* __restrict__ cat) {
  int i = blockIdx.x * 256 + threadIdx.x;
  if (i >= N_NODES * HID) return;
  int n = i >> 8, c = i & 255;
  float acc = b[c];
  const float* xr = x + n * 10;
  #pragma unroll
  for (int j = 0; j < 10; ++j) acc += xr[j] * W[j * HID + c];
  acc = fmaxf(acc, 0.f);
  h[i] = acc;
  h16[i] = (_Float16)acc;
  cat[(size_t)n * 512 + c] = (_Float16)acc;
}

// wave per edge: attn[e][h] = leaky(dot(Q[dst],K[src])*scale + edge_attr@We)
// plus block-level global max -> encoded atomicMax
__global__ __launch_bounds__(256) void k_edge_attn(
    const float* __restrict__ Q, const float* __restrict__ Km,
    const float* __restrict__ ea, const float* __restrict__ We,
    const long long* __restrict__ srcI, const long long* __restrict__ dstI,
    float* __restrict__ attn, unsigned* __restrict__ gmax)
{
  __shared__ float red[8];
  const int e  = blockIdx.x * 8 + (threadIdx.x >> 5);
  const int ln = threadIdx.x & 31;
  float a8 = -3.4e38f;
  if (e < N_EDGES) {
    const long long s = srcI[e], d = dstI[e];
    const float* qr = Q  + (size_t)d * HID + ln * 8;
    const float* kr = Km + (size_t)s * HID + ln * 8;
    float4 q0 = *(const float4*)qr, q1 = *(const float4*)(qr + 4);
    float4 k0 = *(const float4*)kr, k1 = *(const float4*)(kr + 4);
    float dot = q0.x*k0.x + q0.y*k0.y + q0.z*k0.z + q0.w*k0.w
              + q1.x*k1.x + q1.y*k1.y + q1.z*k1.z + q1.w*k1.w;
    dot += __shfl_xor(dot, 1, 32);
    dot += __shfl_xor(dot, 2, 32);            // head = 4 lanes (32 chans)
    if ((ln & 3) == 0) {
      const int hh = ln >> 2;
      float bias = ea[(size_t)e*3+0]*We[0*NHEAD+hh]
                 + ea[(size_t)e*3+1]*We[1*NHEAD+hh]
                 + ea[(size_t)e*3+2]*We[2*NHEAD+hh];
      float v = dot * 0.17677669529663687f + bias;   // 1/sqrt(32)
      v = (v > 0.f) ? v : 0.2f * v;                  // leaky_relu
      attn[(size_t)e * NHEAD + hh] = v;
      a8 = v;
    }
  }
  #pragma unroll
  for (int m = 16; m >= 1; m >>= 1) a8 = fmaxf(a8, __shfl_xor(a8, m, 32));
  if (ln == 0) red[threadIdx.x >> 5] = a8;
  __syncthreads();
  if (threadIdx.x < 8) {
    float v = red[threadIdx.x];
    v = fmaxf(v, __shfl_xor(v, 4, 32));
    v = fmaxf(v, __shfl_xor(v, 2, 32));
    v = fmaxf(v, __shfl_xor(v, 1, 32));
    if (threadIdx.x == 0) atomicMax(gmax, encF(v));
  }
}

// in-place: attn[i] = exp(attn[i]-max); segment_sum over dst
__global__ void k_exp_sum(float* __restrict__ attn, float* __restrict__ asum,
                          const long long* __restrict__ dstI,
                          const unsigned* __restrict__ gmax) {
  int i = blockIdx.x * 256 + threadIdx.x;
  if (i >= N_EDGES * NHEAD) return;
  const float m = decF(*gmax);
  float v = expf(attn[i] - m);
  attn[i] = v;
  const long long d = dstI[i >> 3];
  atomAddF(&asum[(size_t)d * NHEAD + (i & 7)], v);
}

// wave per edge: agg[dst] += V[src] * attn_exp/clip(attn_sum[dst])
__global__ __launch_bounds__(256) void k_scatter(
    const float* __restrict__ V, const float* __restrict__ attn,
    const float* __restrict__ asum, const long long* __restrict__ srcI,
    const long long* __restrict__ dstI, float* __restrict__ agg)
{
  const int e  = blockIdx.x * 8 + (threadIdx.x >> 5);
  const int ln = threadIdx.x & 31;
  if (e >= N_EDGES) return;
  const long long s = srcI[e], d = dstI[e];
  const int hh = ln >> 2;
  const float sum = fmaxf(asum[(size_t)d * NHEAD + hh], 1e-12f);
  const float wgt = attn[(size_t)e * NHEAD + hh] / sum;
  const float* vr = V + (size_t)s * HID + ln * 8;
  float* ar = agg + (size_t)d * HID + ln * 8;
  float4 v0 = *(const float4*)vr, v1 = *(const float4*)(vr + 4);
  atomAddF(ar + 0, v0.x * wgt); atomAddF(ar + 1, v0.y * wgt);
  atomAddF(ar + 2, v0.z * wgt); atomAddF(ar + 3, v0.w * wgt);
  atomAddF(ar + 4, v1.x * wgt); atomAddF(ar + 5, v1.y * wgt);
  atomAddF(ar + 6, v1.z * wgt); atomAddF(ar + 7, v1.w * wgt);
}

// wave per node: h = LN(h + upd)*gamma + beta; refresh f16 mirrors
__global__ __launch_bounds__(256) void k_ln(
    float* __restrict__ h, const float* __restrict__ upd,
    const float* __restrict__ gamma, const float* __restrict__ beta,
    _Float16* __restrict__ h16, _Float16* __restrict__ cat)
{
  const int n  = blockIdx.x * 8 + (threadIdx.x >> 5);
  const int ln = threadIdx.x & 31;
  if (n >= N_NODES) return;
  const size_t base = (size_t)n * HID + ln * 8;
  float t[8]; float s = 0.f, s2 = 0.f;
  #pragma unroll
  for (int j = 0; j < 8; ++j) {
    t[j] = h[base + j] + upd[base + j];
    s += t[j]; s2 += t[j] * t[j];
  }
  #pragma unroll
  for (int m = 16; m >= 1; m >>= 1) { s += __shfl_xor(s, m, 32); s2 += __shfl_xor(s2, m, 32); }
  const float mu  = s * (1.f / HID);
  const float var = s2 * (1.f / HID) - mu * mu;
  const float r   = rsqrtf(var + 1e-5f);
  #pragma unroll
  for (int j = 0; j < 8; ++j) {
    const int c = ln * 8 + j;
    const float o = (t[j] - mu) * r * gamma[c] + beta[c];
    h[base + j] = o;
    h16[base + j] = (_Float16)o;
    cat[(size_t)n * 512 + c] = (_Float16)o;
  }
}

// wave per node: out[n] = dot(h1[n], W2) + b2
__global__ __launch_bounds__(256) void k_head2(
    const float* __restrict__ h1, const float* __restrict__ W2,
    const float* __restrict__ b2, float* __restrict__ out)
{
  const int n  = blockIdx.x * 8 + (threadIdx.x >> 5);
  const int ln = threadIdx.x & 31;
  if (n >= N_NODES) return;
  const float* r = h1 + (size_t)n * 128;
  float acc = r[ln] * W2[ln] + r[ln + 32] * W2[ln + 32]
            + r[ln + 64] * W2[ln + 64] + r[ln + 96] * W2[ln + 96];
  #pragma unroll
  for (int m = 16; m >= 1; m >>= 1) acc += __shfl_xor(acc, m, 32);
  if (ln == 0) out[n] = acc + b2[0];
}

// ---------------------------------------------------------------------------
extern "C" void kernel_launch(void* const* d_in, const int* in_sizes, int n_in,
                              void* d_out, int out_size, void* d_ws, size_t ws_size,
                              hipStream_t stream) {
  (void)in_sizes; (void)n_in; (void)out_size; (void)ws_size;
  const float* x    = (const float*)d_in[0];
  const float* ea   = (const float*)d_in[1];
  const float* Win  = (const float*)d_in[2];
  const float* bin  = (const float*)d_in[3];
  const float* Wq   = (const float*)d_in[4];
  const float* Wk   = (const float*)d_in[5];
  const float* Wv   = (const float*)d_in[6];
  const float* We   = (const float*)d_in[7];
  const float* Wo   = (const float*)d_in[8];
  const float* bo   = (const float*)d_in[9];
  const float* Wm   = (const float*)d_in[10];
  const float* bm   = (const float*)d_in[11];
  const float* gm   = (const float*)d_in[12];
  const float* bt   = (const float*)d_in[13];
  const float* Wh1  = (const float*)d_in[14];
  const float* bh1  = (const float*)d_in[15];
  const float* Wh2  = (const float*)d_in[16];
  const float* bh2  = (const float*)d_in[17];
  const long long* eidx = (const long long*)d_in[18];
  const long long* srcI = eidx;
  const long long* dstI = eidx + N_EDGES;
  float* outp = (float*)d_out;

  const size_t NHf = (size_t)N_NODES * HID;
  char* p = (char*)d_ws;
  auto alloc = [&](size_t bytes) -> void* {
    void* r = (void*)p; p += (bytes + 255) & ~(size_t)255; return r;
  };
  float*    hF     = (float*)alloc(NHf * 4);
  float*    Qf     = (float*)alloc(NHf * 4);   // reused: agg, head1
  float*    Kf     = (float*)alloc(NHf * 4);   // reused: agg@Wo
  float*    Vf     = (float*)alloc(NHf * 4);   // reused: upd
  _Float16* h16    = (_Float16*)alloc(NHf * 2);
  _Float16* agg16  = (_Float16*)alloc(NHf * 2);
  _Float16* cat16  = (_Float16*)alloc((size_t)N_NODES * 512 * 2);
  float*    attn   = (float*)alloc((size_t)N_EDGES * NHEAD * 4);
  float*    asum   = (float*)alloc((size_t)N_NODES * NHEAD * 4);
  unsigned* gmax   = (unsigned*)alloc(256);
  _Float16* wq16   = (_Float16*)alloc((size_t)NLAYER * HID * HID * 2);
  _Float16* wk16   = (_Float16*)alloc((size_t)NLAYER * HID * HID * 2);
  _Float16* wv16   = (_Float16*)alloc((size_t)NLAYER * HID * HID * 2);
  _Float16* wo16   = (_Float16*)alloc((size_t)NLAYER * HID * HID * 2);
  _Float16* wm16   = (_Float16*)alloc((size_t)NLAYER * 512 * HID * 2);
  _Float16* wh116  = (_Float16*)alloc((size_t)HID * 128 * 2);

  const int TPB = 256;
  const int LHH = NLAYER * HID * HID;        // 196608
  const int LWM = NLAYER * 512 * HID;        // 393216
  auto cdiv = [](int a, int b) { return (a + b - 1) / b; };

  // weight conversion f32 -> f16 (cheap; every launch for determinism)
  k_cvt<<<cdiv(LHH, TPB), TPB, 0, stream>>>(Wq,  wq16,  LHH);
  k_cvt<<<cdiv(LHH, TPB), TPB, 0, stream>>>(Wk,  wk16,  LHH);
  k_cvt<<<cdiv(LHH, TPB), TPB, 0, stream>>>(Wv,  wv16,  LHH);
  k_cvt<<<cdiv(LHH, TPB), TPB, 0, stream>>>(Wo,  wo16,  LHH);
  k_cvt<<<cdiv(LWM, TPB), TPB, 0, stream>>>(Wm,  wm16,  LWM);
  k_cvt<<<cdiv(HID * 128, TPB), TPB, 0, stream>>>(Wh1, wh116, HID * 128);

  // input projection
  k_inproj<<<cdiv(N_NODES * HID, TPB), TPB, 0, stream>>>(x, Win, bin, hF, h16, cat16);

  const dim3 gemmHH(cdiv(N_NODES, 128), HID / 64);    // 391 x 4
  const dim3 gemmH1(cdiv(N_NODES, 128), 128 / 64);    // 391 x 2
  const int  gridE8 = cdiv(N_EDGES, 8);
  const int  gridN8 = cdiv(N_NODES, 8);

  for (int l = 0; l < NLAYER; ++l) {
    const _Float16* wq = wq16 + (size_t)l * HID * HID;
    const _Float16* wk = wk16 + (size_t)l * HID * HID;
    const _Float16* wv = wv16 + (size_t)l * HID * HID;
    const _Float16* wo = wo16 + (size_t)l * HID * HID;
    const _Float16* wm = wm16 + (size_t)l * 512 * HID;

    // Q, K, V projections (no bias)
    k_gemm<0, false><<<gemmHH, TPB, 0, stream>>>(h16, wq, nullptr, Qf, nullptr, N_NODES, HID, HID, 0);
    k_gemm<0, false><<<gemmHH, TPB, 0, stream>>>(h16, wk, nullptr, Kf, nullptr, N_NODES, HID, HID, 0);
    k_gemm<0, false><<<gemmHH, TPB, 0, stream>>>(h16, wv, nullptr, Vf, nullptr, N_NODES, HID, HID, 0);

    // softmax accumulators
    k_init_small<<<cdiv(N_NODES * NHEAD, TPB), TPB, 0, stream>>>(asum, gmax, N_NODES * NHEAD);
    // attention logits + global max
    k_edge_attn<<<gridE8, TPB, 0, stream>>>(Qf, Kf, ea, We + (size_t)l * 3 * NHEAD,
                                            srcI, dstI, attn, gmax);
    // Q dead -> reuse as agg
    k_zero<<<cdiv((int)NHf, TPB), TPB, 0, stream>>>(Qf, (int)NHf);
    // exp + segment sum
    k_exp_sum<<<cdiv(N_EDGES * NHEAD, TPB), TPB, 0, stream>>>(attn, asum, dstI, gmax);
    // weighted scatter of V into agg(=Qf)
    k_scatter<<<gridE8, TPB, 0, stream>>>(Vf, attn, asum, srcI, dstI, Qf);

    // agg -> f16, then agg@Wo + bo  (f32 into Kf, f16 into concat 2nd half)
    k_cvt<<<cdiv((int)NHf, TPB), TPB, 0, stream>>>(Qf, agg16, (int)NHf);
    k_gemm<0, true><<<gemmHH, TPB, 0, stream>>>(agg16, wo, bo + (size_t)l * HID,
                                                Kf, cat16 + HID, N_NODES, HID, HID, 512);
    // upd = relu(concat @ Wm + bm) into Vf
    k_gemm<1, false><<<gemmHH, TPB, 0, stream>>>(cat16, wm, bm + (size_t)l * HID,
                                                 Vf, nullptr, N_NODES, 512, HID, 0);
    // h = LN(h + upd); refresh mirrors
    k_ln<<<gridN8, TPB, 0, stream>>>(hF, Vf, gm + (size_t)l * HID, bt + (size_t)l * HID,
                                     h16, cat16);
  }

  // output head: relu(h @ Wh1 + bh1) -> Qf (N x 128), then dot with Wh2
  k_gemm<1, false><<<gemmH1, TPB, 0, stream>>>(h16, wh116, bh1, Qf, nullptr, N_NODES, HID, 128, 0);
  k_head2<<<gridN8, TPB, 0, stream>>>(Qf, Wh2, bh2, outp);
}
